// RNNWithAttention_56066503082250
// MI455X (gfx1250) — compile-verified, matrix operand order checked by reading, not run on hardware
//
#include <hip/hip_runtime.h>
#include <hip/hip_bf16.h>

typedef __attribute__((ext_vector_type(16))) __bf16 v16bf;
typedef __attribute__((ext_vector_type(8)))  __bf16 v8bf;
typedef __attribute__((ext_vector_type(8)))  float  v8f;
typedef unsigned int __attribute__((ext_vector_type(4))) u32x4;
typedef int __attribute__((ext_vector_type(4))) i32x4;
typedef int __attribute__((ext_vector_type(8))) i32x8;

#define Bc 8
#define Tc 512
#define Vc 32000
#define Ec 256
#define Hc 512

// ---------- WMMA helpers (CDNA5 16x16x32 bf16, f32 accum) ----------

__device__ __forceinline__ v8f wmma_bf16(v16bf a, v16bf b, v8f c) {
  return __builtin_amdgcn_wmma_f32_16x16x32_bf16(
      /*neg_a=*/false, a, /*neg_b=*/false, b,
      /*c_mod=*/(short)0, c, /*reuse_a=*/false, /*reuse_b=*/false);
}

// Load a 16x32 bf16 fragment from a row-major bf16 matrix (global or LDS).
// rowbase points at element [row 0][k0]; ld = row stride in elements.
// Lane mapping per ISA 7.12.2: row = lane&15; K halves at kb and 16+kb.
__device__ __forceinline__ v16bf frag_from_bf(const __bf16* rowbase, int ld) {
  const int lane = threadIdx.x & 31;
  const int r  = lane & 15;
  const int kb = (lane >> 4) << 3;
  const __bf16* p = rowbase + (long)r * ld + kb;
  v8bf lo = *(const v8bf*)(p);
  v8bf hi = *(const v8bf*)(p + 16);
  v16bf f;
#pragma unroll
  for (int i = 0; i < 8; ++i) { f[i] = lo[i]; f[8 + i] = hi[i]; }
  return f;
}

// Same, but source is f32 row-major (converted on the fly).
__device__ __forceinline__ v16bf frag_from_f32(const float* rowbase, int ld) {
  const int lane = threadIdx.x & 31;
  const int r  = lane & 15;
  const int kb = (lane >> 4) << 3;
  const float* p = rowbase + (long)r * ld + kb;
  v8f lo = *(const v8f*)(p);
  v8f hi = *(const v8f*)(p + 16);
  v16bf f;
#pragma unroll
  for (int i = 0; i < 8; ++i) { f[i] = (__bf16)lo[i]; f[8 + i] = (__bf16)hi[i]; }
  return f;
}

// ---------- Tensor Data Mover: 2D bf16 tile -> LDS (ISA ch.8 D# layout) ----------
// tile (tile1 rows x tile0 elems, contiguous dim0), row stride = stride0 elements.

__device__ __forceinline__ void tdm_load_2d_bf16(unsigned int lds_off, const void* gsrc,
                                                 unsigned int dim0, unsigned int dim1,
                                                 unsigned int tile0, unsigned int tile1,
                                                 unsigned long long stride0) {
  unsigned long long ga = (unsigned long long)(uintptr_t)gsrc;
  u32x4 g0;
  g0[0] = 1u;                                              // count=1, user mode
  g0[1] = lds_off;                                         // lds_addr (bytes)
  g0[2] = (unsigned int)(ga & 0xffffffffu);                // global_addr[31:0]
  g0[3] = (unsigned int)((ga >> 32) & 0x01ffffffu) | (2u << 30);  // [120:96] + type=2
  i32x8 g1;
  g1[0] = (int)(1u << 16);                                 // data_size=1 (2 bytes), no mcast
  g1[1] = (int)((dim0 & 0xffffu) << 16);                   // tensor_dim0[15:0] @ bits 63:48
  g1[2] = (int)((dim0 >> 16) | ((dim1 & 0xffffu) << 16));  // dim0[31:16] | dim1[15:0]
  g1[3] = (int)((dim1 >> 16) | (tile0 << 16));             // dim1[31:16] | tile_dim0
  g1[4] = (int)(tile1 & 0xffffu);                          // tile_dim1 (tile_dim2=0)
  g1[5] = (int)(unsigned int)(stride0 & 0xffffffffu);      // tensor_dim0_stride[31:0]
  g1[6] = (int)(unsigned int)((stride0 >> 32) & 0xffffu);  // stride0[47:32], stride1=0
  g1[7] = 0;
  i32x4 z4 = {0, 0, 0, 0};
#if defined(__clang_major__) && __clang_major__ >= 23
  i32x8 z8 = {0, 0, 0, 0, 0, 0, 0, 0};
  __builtin_amdgcn_tensor_load_to_lds(g0, g1, z4, z4, z8, 0);
#else
  __builtin_amdgcn_tensor_load_to_lds(g0, g1, z4, z4, 0);
#endif
}

// ---------- generic f32 -> bf16 convert ----------

__global__ void k_cvt_bf16(const float* __restrict__ in, __bf16* __restrict__ out, long n) {
  long i = (long)blockIdx.x * blockDim.x + threadIdx.x;
  if (i < n) out[i] = (__bf16)in[i];
}

// ---------- kernel 1: embedding gather + input projection (xin = emb @ W_ih^T + b_ih) ----------
// grid (4096/16, 512/64), block 128 (4 waves); wave w owns N tile n0 = by*64 + 16w.

__global__ void k_embed_proj(const int* __restrict__ x, const float* __restrict__ embed,
                             const float* __restrict__ W_ih, const float* __restrict__ b_ih,
                             float* __restrict__ xin) {
  const int lane = threadIdx.x & 31;
  const int wave = threadIdx.x >> 5;
  const int m0 = blockIdx.x * 16;
  const int n0 = blockIdx.y * 64 + wave * 16;
  const int r  = lane & 15;
  const int kb = (lane >> 4) << 3;
  const int bt = m0 + r;
  const int tok = x[bt];
  const float scale = tok ? 1.f : 0.f;        // row 0 of the table is forced to zero
  const float* arow = embed + (long)tok * Ec;

  v8f acc = {};
#pragma unroll
  for (int k0 = 0; k0 < Ec; k0 += 32) {
    v8f lo = *(const v8f*)(arow + k0 + kb);       // branch-free gather, 32B vector loads
    v8f hi = *(const v8f*)(arow + k0 + kb + 16);
    v16bf a;
#pragma unroll
    for (int i = 0; i < 8; ++i) {
      a[i]     = (__bf16)(lo[i] * scale);
      a[8 + i] = (__bf16)(hi[i] * scale);
    }
    v16bf b = frag_from_f32(W_ih + (long)n0 * Ec + k0, Ec);
    acc = wmma_bf16(a, b, acc);
  }
  const int n   = n0 + (lane & 15);
  const int mm0 = (lane >> 4) << 3;
  const float bias = b_ih[n];
#pragma unroll
  for (int i = 0; i < 8; ++i)
    xin[(long)(m0 + mm0 + i) * Hc + n] = acc[i] + bias;
}

// ---------- kernel 2: serial RNN scan, one workgroup (32 waves), h tile in LDS ----------
// h kept as 16x512 bf16 (rows 8..15 zero). Wave w computes N slice n0=16w each step.

__global__ void k_rnn_scan(const float* __restrict__ xin, const __bf16* __restrict__ Whh_bf,
                           const float* __restrict__ b_hh, const int* __restrict__ lengths,
                           float* __restrict__ outbuf, __bf16* __restrict__ feats_bf) {
  __shared__ __align__(16) __bf16 hbf[16 * Hc];
  const int lane = threadIdx.x & 31;
  const int wave = threadIdx.x >> 5;        // 0..31
  const int n0 = wave * 16;
  const int n  = n0 + (lane & 15);
  const float bh = b_hh[n];

  for (int i = threadIdx.x; i < 16 * Hc; i += blockDim.x) hbf[i] = (__bf16)0.f;
  __syncthreads();

  for (int t = 0; t < Tc; ++t) {
    v8f acc = {};
#pragma unroll 4
    for (int k0 = 0; k0 < Hc; k0 += 32) {
      v16bf a = frag_from_bf(hbf + k0, Hc);                       // LDS (ds_load_b128)
      v16bf b = frag_from_bf(Whh_bf + (long)n0 * Hc + k0, Hc);    // L2-resident weights
      acc = wmma_bf16(a, b, acc);
    }
    __syncthreads();                          // all waves done reading h(t-1)
    if (lane < 16) {                          // lanes 0..15 carry rows m=0..7 (the real batch)
#pragma unroll
      for (int i = 0; i < 8; ++i) {
        const int m = i;                      // batch index
        const long bt = (long)m * Tc + t;
        float v = tanhf(xin[bt * Hc + n] + acc[i] + bh);
        hbf[m * Hc + n] = (__bf16)v;          // carry state (unmasked, as in reference)
        float mv = (t < lengths[m]) ? v : 0.f;
        outbuf[bt * Hc + n] = mv;             // masked output, f32
        feats_bf[bt * (2 * Hc) + n] = (__bf16)mv;  // masked output, bf16 (feats[:, :H])
      }
    }
    __syncthreads();                          // h(t) visible before next step
  }
}

// ---------- kernel 3: q / k projections over masked out ----------
// grid (256, 8, 2): z=0 -> q = out@W1^T ; z=1 -> k = out@W2^T + attn_b

__global__ void k_qk(const __bf16* __restrict__ feats_bf, const __bf16* __restrict__ attnW_bf,
                     const float* __restrict__ attn_b, float* __restrict__ qbuf,
                     float* __restrict__ kbuf) {
  const int lane = threadIdx.x & 31;
  const int wave = threadIdx.x >> 5;
  const int m0 = blockIdx.x * 16;
  const int n0 = blockIdx.y * 64 + wave * 16;
  const int which = blockIdx.z;                  // 0 = W1 (cols 0..511), 1 = W2 (cols 512..1023)
  const __bf16* B0 = attnW_bf + which * Hc;

  v8f acc = {};
#pragma unroll 4
  for (int k0 = 0; k0 < Hc; k0 += 32) {
    __builtin_prefetch(B0 + (long)n0 * (2 * Hc) + k0 + 128, 0, 1);  // global_prefetch
    v16bf a = frag_from_bf(feats_bf + (long)m0 * (2 * Hc) + k0, 2 * Hc);
    v16bf b = frag_from_bf(B0 + (long)n0 * (2 * Hc) + k0, 2 * Hc);
    acc = wmma_bf16(a, b, acc);
  }
  const int n   = n0 + (lane & 15);
  const int mm0 = (lane >> 4) << 3;
  const float bias = which ? attn_b[n] : 0.f;
  float* dst = which ? kbuf : qbuf;
#pragma unroll
  for (int i = 0; i < 8; ++i)
    dst[(long)(m0 + mm0 + i) * Hc + n] = acc[i] + bias;
}

// ---------- kernel 4: additive attention scores + softmax + context (VALU-bound) ----------
// One block per (b,t). scores[s] = v . tanh(q[b,t]+k[b,s]) + vb over valid s < min(t+1,len).

__global__ void k_attn(const float* __restrict__ qbuf, const float* __restrict__ kbuf,
                       const float* __restrict__ outbuf, const float* __restrict__ attn_v,
                       const float* __restrict__ attn_v_b, const int* __restrict__ lengths,
                       __bf16* __restrict__ feats_bf) {
  __shared__ float qs[Hc];
  __shared__ float vs[Hc];
  __shared__ float sc[Tc];
  __shared__ float red[256];

  const int bt = blockIdx.x;
  const int b  = bt >> 9;
  const int t  = bt & (Tc - 1);
  const int tid = threadIdx.x;

  for (int i = tid; i < Hc; i += 256) { qs[i] = qbuf[(long)bt * Hc + i]; vs[i] = attn_v[i]; }
  __syncthreads();

  const int len  = lengths[b];
  const int slim = (t + 1 < len) ? (t + 1) : len;   // valid range; rest gets alpha = 0
  const float vb = attn_v_b[0];

  for (int s = tid; s < slim; s += 256) {
    const float* krow = kbuf + ((long)b * Tc + s) * Hc;
    float acc = vb;
#pragma unroll 4
    for (int h = 0; h < Hc; ++h) acc += vs[h] * tanhf(qs[h] + krow[h]);
    sc[s] = acc;
  }
  __syncthreads();

  // masked softmax over sc[0..slim)
  float m = -1e30f;
  for (int s = tid; s < slim; s += 256) m = fmaxf(m, sc[s]);
  red[tid] = m; __syncthreads();
  for (int o = 128; o > 0; o >>= 1) { if (tid < o) red[tid] = fmaxf(red[tid], red[tid + o]); __syncthreads(); }
  m = red[0]; __syncthreads();

  float sum = 0.f;
  for (int s = tid; s < slim; s += 256) { float e = __expf(sc[s] - m); sc[s] = e; sum += e; }
  red[tid] = sum; __syncthreads();
  for (int o = 128; o > 0; o >>= 1) { if (tid < o) red[tid] += red[tid + o]; __syncthreads(); }
  const float inv = 1.f / red[0]; __syncthreads();
  for (int s = tid; s < slim; s += 256) sc[s] *= inv;
  __syncthreads();

  // context[h] = sum_s alpha[s] * out[b,s,h]  -> feats[:, H:2H]
  for (int h = tid; h < Hc; h += 256) {
    float acc = 0.f;
    for (int s = 0; s < slim; ++s) acc += sc[s] * outbuf[((long)b * Tc + s) * Hc + h];
    feats_bf[(long)bt * (2 * Hc) + Hc + h] = (__bf16)acc;
  }
}

// ---------- kernel 5: logits = feats @ fc_W^T + fc_b (dominant GEMM, 268 GFLOP) ----------
// 256 threads (8 waves), C tile 32x128. TDM double-buffers A (32x32) and B (128x32)
// bf16 tiles into LDS each K-step; waves read fragments via ds_load_b128.

__global__ void k_logits(const __bf16* __restrict__ feats_bf, const __bf16* __restrict__ fcW_bf,
                         const float* __restrict__ fc_b, float* __restrict__ logits) {
  __shared__ __align__(16) __bf16 Abuf[2][32 * 32];
  __shared__ __align__(16) __bf16 Bbuf[2][128 * 32];
  const int lane = threadIdx.x & 31;
  const int wave = threadIdx.x >> 5;           // 0..7
  const int m0 = blockIdx.x * 32;
  const int n0 = blockIdx.y * 128;
  const int KT = (2 * Hc) / 32;                // 32 K-steps

  v8f acc0 = {}, acc1 = {};

  if (wave == 0) {
    tdm_load_2d_bf16((unsigned int)(uintptr_t)&Abuf[0][0], feats_bf + (long)m0 * (2 * Hc),
                     2 * Hc, Bc * Tc, 32, 32, 2 * Hc);
    tdm_load_2d_bf16((unsigned int)(uintptr_t)&Bbuf[0][0], fcW_bf + (long)n0 * (2 * Hc),
                     2 * Hc, Vc, 32, 128, 2 * Hc);
  }

  for (int k = 0; k < KT; ++k) {
    if (wave == 0) __builtin_amdgcn_s_wait_tensorcnt(0);   // current buffer landed in LDS
    __syncthreads();
    const int cur = k & 1;
    if (wave == 0 && (k + 1) < KT) {                       // prefetch next tiles via TDM
      const int nxt = (k + 1) & 1;
      tdm_load_2d_bf16((unsigned int)(uintptr_t)&Abuf[nxt][0],
                       feats_bf + (long)m0 * (2 * Hc) + (k + 1) * 32,
                       2 * Hc, Bc * Tc, 32, 32, 2 * Hc);
      tdm_load_2d_bf16((unsigned int)(uintptr_t)&Bbuf[nxt][0],
                       fcW_bf + (long)n0 * (2 * Hc) + (k + 1) * 32,
                       2 * Hc, Vc, 32, 128, 2 * Hc);
    }
    v16bf a0 = frag_from_bf(&Abuf[cur][0], 32);
    v16bf a1 = frag_from_bf(&Abuf[cur][16 * 32], 32);
    v16bf b  = frag_from_bf(&Bbuf[cur][wave * 16 * 32], 32);
    acc0 = wmma_bf16(a0, b, acc0);
    acc1 = wmma_bf16(a1, b, acc1);                         // B fragment reused: 2x intensity
    __syncthreads();                                       // done reading before overwrite
  }

  const int n   = n0 + wave * 16 + (lane & 15);
  const int mm0 = (lane >> 4) << 3;
  const float bias = fc_b[n];
#pragma unroll
  for (int i = 0; i < 8; ++i) {
    logits[(long)(m0 + mm0 + i) * Vc + n]      = acc0[i] + bias;
    logits[(long)(m0 + 16 + mm0 + i) * Vc + n] = acc1[i] + bias;
  }
}

// ---------- host launcher ----------

extern "C" void kernel_launch(void* const* d_in, const int* in_sizes, int n_in,
                              void* d_out, int out_size, void* d_ws, size_t ws_size,
                              hipStream_t stream) {
  const int*   x        = (const int*)d_in[0];
  const int*   lengths  = (const int*)d_in[1];
  const float* embed    = (const float*)d_in[2];
  const float* W_ih     = (const float*)d_in[3];
  const float* W_hh     = (const float*)d_in[4];
  const float* b_ih     = (const float*)d_in[5];
  const float* b_hh     = (const float*)d_in[6];
  const float* attn_W   = (const float*)d_in[7];
  const float* attn_b   = (const float*)d_in[8];
  const float* attn_v   = (const float*)d_in[9];
  const float* attn_v_b = (const float*)d_in[10];
  const float* fc_W     = (const float*)d_in[11];
  const float* fc_b     = (const float*)d_in[12];
  float* logits = (float*)d_out;

  char* ws = (char*)d_ws;
  size_t off = 0;
  auto salloc = [&](size_t bytes) -> void* {
    void* p = ws + off;
    off += (bytes + 255) & ~(size_t)255;
    return p;
  };
  __bf16* Whh_bf   = (__bf16*)salloc((size_t)Hc * Hc * 2);
  __bf16* attnW_bf = (__bf16*)salloc((size_t)Hc * 2 * Hc * 2);
  __bf16* fcW_bf   = (__bf16*)salloc((size_t)Vc * 2 * Hc * 2);
  float*  xin      = (float*)salloc((size_t)Bc * Tc * Hc * 4);
  float*  outbuf   = (float*)salloc((size_t)Bc * Tc * Hc * 4);
  float*  qbuf     = (float*)salloc((size_t)Bc * Tc * Hc * 4);
  float*  kbuf     = (float*)salloc((size_t)Bc * Tc * Hc * 4);
  __bf16* feats_bf = (__bf16*)salloc((size_t)Bc * Tc * 2 * Hc * 2);

  // weight conversions (once per call; L2-resident afterwards)
  {
    long n1 = (long)Hc * Hc;
    long n2 = (long)Hc * 2 * Hc;
    long n3 = (long)Vc * 2 * Hc;
    k_cvt_bf16<<<dim3((unsigned)((n1 + 255) / 256)), 256, 0, stream>>>(W_hh, Whh_bf, n1);
    k_cvt_bf16<<<dim3((unsigned)((n2 + 255) / 256)), 256, 0, stream>>>(attn_W, attnW_bf, n2);
    k_cvt_bf16<<<dim3((unsigned)((n3 + 255) / 256)), 256, 0, stream>>>(fc_W, fcW_bf, n3);
  }

  k_embed_proj<<<dim3(Bc * Tc / 16, Hc / 64), 128, 0, stream>>>(x, embed, W_ih, b_ih, xin);
  k_rnn_scan<<<1, 1024, 0, stream>>>(xin, Whh_bf, b_hh, lengths, outbuf, feats_bf);
  k_qk<<<dim3(Bc * Tc / 16, Hc / 64, 2), 128, 0, stream>>>(feats_bf, attnW_bf, attn_b, qbuf, kbuf);
  k_attn<<<dim3(Bc * Tc), 256, 0, stream>>>(qbuf, kbuf, outbuf, attn_v, attn_v_b, lengths, feats_bf);
  k_logits<<<dim3(Bc * Tc / 32, Vc / 128), 256, 0, stream>>>(feats_bf, fcW_bf, fc_b, logits);
}
